// DWTLinearAttention_87840671138276
// MI455X (gfx1250) — compile-verified, matrix operand order checked by reading, not run on hardware
//
#include <hip/hip_runtime.h>

typedef __bf16 bf16_t;
typedef __attribute__((ext_vector_type(16))) __bf16 v16bf;
typedef __attribute__((ext_vector_type(8)))  __bf16 v8bf;
typedef __attribute__((ext_vector_type(8)))  float  v8f;

// Problem constants (from reference setup_inputs)
#define BB   4
#define CC   512
#define MM   64
#define NN   16384      // W*H = 128*128
#define NP   4096       // LL pixels = 64*64
#define WFULL 128

// ---------------------------------------------------------------------------
// DWT LL band:  xl_bf[b][c][pix] (bf16, C x n layout for GEMM B operand)
//               llT[b][pix][c]  (f32, transposed for coalesced final pass)
// ---------------------------------------------------------------------------
__global__ __launch_bounds__(256) void dwt_ll_kernel(
    const float* __restrict__ x, bf16_t* __restrict__ xl_bf,
    float* __restrict__ llT) {
  long long idx = (long long)blockIdx.x * 256 + threadIdx.x;   // B*C*NP
  int pix = (int)(idx & (NP - 1));
  long long t = idx >> 12;
  int c = (int)(t & (CC - 1));
  int b = (int)(t >> 9);
  int i = pix >> 6, j = pix & 63;
  const float* r0 = x + ((long long)b << 23) + ((long long)c << 14)
                      + (long long)(2 * i) * WFULL + 2 * j;
  float a = r0[0], bv_ = r0[1], cv = r0[WFULL], dv = r0[WFULL + 1];
  float ll = 0.5f * (a + bv_ + cv + dv);
  xl_bf[((long long)b << 21) + ((long long)c << 12) + pix] = (bf16_t)ll;
  llT[((long long)b << 21) + ((long long)pix << 9) + c] = ll;
}

// ---------------------------------------------------------------------------
// f32 -> bf16 weight conversion
// ---------------------------------------------------------------------------
__global__ void f32_to_bf16_kernel(const float* __restrict__ src,
                                   bf16_t* __restrict__ dst, int n) {
  int i = blockIdx.x * 256 + threadIdx.x;
  if (i < n) dst[i] = (bf16_t)src[i];
}

// ---------------------------------------------------------------------------
// Column L2 normalize (64 rows x 4096 cols per batch), in place + bf16 copy
// ---------------------------------------------------------------------------
__global__ __launch_bounds__(256) void l2n64_kernel(
    float* __restrict__ Qio, bf16_t* __restrict__ Qbf) {
  int idx = blockIdx.x * 256 + threadIdx.x;    // B*NP
  int pix = idx & (NP - 1);
  int b = idx >> 12;
  float* base = Qio + ((long long)b << 18) + pix;
  bf16_t* bbase = Qbf + ((long long)b << 18) + pix;
  float ss = 0.f;
  #pragma unroll 8
  for (int r = 0; r < MM; ++r) { float v = base[(long long)r << 12]; ss += v * v; }
  float inv = rsqrtf(ss);
  #pragma unroll 8
  for (int r = 0; r < MM; ++r) {
    float v = base[(long long)r << 12] * inv;
    base[(long long)r << 12] = v;
    bbase[(long long)r << 12] = (bf16_t)v;
  }
}

// ---------------------------------------------------------------------------
// Row sums (one block per (batch,row)) with eps fold:  out[b*rows+r]=sum+eps
// ---------------------------------------------------------------------------
__global__ __launch_bounds__(256) void rowsum_kernel(
    const float* __restrict__ src, float* __restrict__ out,
    int cols, float eps, long long sSrc, int rowsPerBatch) {
  int gr = blockIdx.x;
  int b = gr / rowsPerBatch;
  int r = gr - b * rowsPerBatch;
  const float* p = src + (long long)b * sSrc + (long long)r * cols;
  float s = 0.f;
  for (int ci = threadIdx.x; ci < cols; ci += 256) s += p[ci];
  __shared__ float red[256];
  red[threadIdx.x] = s;
  __syncthreads();
  for (int st = 128; st > 0; st >>= 1) {
    if ((int)threadIdx.x < st) red[threadIdx.x] += red[threadIdx.x + st];
    __syncthreads();
  }
  if (threadIdx.x == 0) out[gr] = red[0] + eps;
}

// ---------------------------------------------------------------------------
// tailor[b][pix] = 1/(n + dot(Qn[:,pix], Ksum))
// ---------------------------------------------------------------------------
__global__ __launch_bounds__(256) void tailor_kernel(
    const float* __restrict__ Qn, const float* __restrict__ Ksum,
    float* __restrict__ tailor) {
  int idx = blockIdx.x * 256 + threadIdx.x;    // B*NP
  int pix = idx & (NP - 1);
  int b = idx >> 12;
  const float* qb = Qn + ((long long)b << 18) + pix;
  const float* ks = Ksum + b * MM;
  float dot = 0.f;
  #pragma unroll 8
  for (int r = 0; r < MM; ++r) dot += qb[(long long)r << 12] * ks[r];
  tailor[idx] = 1.0f / ((float)NP + dot);
}

// ---------------------------------------------------------------------------
// bf16 WMMA GEMM: D[M,N] = alpha*rowScale[m]*((A@B)[m,n]+biasRow[m]+biasCol[n])
// Block tile 64x128, 8 waves of 32x32 (2x2 WMMA 16x16x32 bf16 tiles).
// LDS tiles are both K-contiguous per row (As: M-major, Bt: N-major) so every
// fragment is two 16-byte ds_load_b128 ops.
// ATR: A stored K-major (KxM);  BTR: B stored N-major (NxK).
// ---------------------------------------------------------------------------
template <bool ATR, bool BTR>
__global__ __launch_bounds__(256) void gemm_bf16_wmma(
    const bf16_t* __restrict__ A, long long lda, long long sA,
    const bf16_t* __restrict__ Bm, long long ldb, long long sB,
    float* __restrict__ D, long long ldd, long long sD,
    bf16_t* __restrict__ Dbf,
    const float* __restrict__ biasRow, long long sBR,
    const float* __restrict__ biasCol, long long sBC,
    const float* __restrict__ rowScale, long long sRS,
    const float* __restrict__ alphaPtr,
    int M, int N, int K) {
  __shared__ __align__(16) bf16_t As[64][40];    // [m][k], k contiguous
  __shared__ __align__(16) bf16_t Bt[128][40];   // [n][k], k contiguous

  const int b  = blockIdx.z;
  const int n0 = blockIdx.x * 128;
  const int m0 = blockIdx.y * 64;
  const bf16_t* Ab = A + (long long)b * sA;
  const bf16_t* Bb = Bm + (long long)b * sB;

  const int tid  = threadIdx.x;
  const int lane = tid & 31;
  const int wid  = tid >> 5;
  const int wm   = wid >> 2;   // 0..1
  const int wn   = wid & 3;    // 0..3
  const int half = lane >> 4;  // 0/1
  const int r16  = lane & 15;

  // ---- per-thread staging descriptors (one v8bf for A, two for B) ----
  const bf16_t* ag;
  long long astep;
  bf16_t* a_st;
  if constexpr (!ATR) {
    const int ml = tid >> 2, kseg = (tid & 3) * 8;
    ag = Ab + (long long)(m0 + ml) * lda + kseg;
    astep = 32;
    a_st = &As[ml][kseg];                 // contiguous 16B store
  } else {
    const int kl = tid >> 3, mseg = (tid & 7) * 8;
    ag = Ab + (long long)kl * lda + (m0 + mseg);
    astep = 32 * lda;
    a_st = &As[mseg][kl];                 // scatter store, stride 40
  }
  const bf16_t *bg0, *bg1;
  bf16_t *b_st0, *b_st1;
  long long bstep;
  if constexpr (!BTR) {
    const int i1 = 256 + tid;
    const int kl0 = tid >> 4, ns0 = (tid & 15) * 8;
    const int kl1 = i1 >> 4,  ns1 = (i1 & 15) * 8;
    bg0 = Bb + (long long)kl0 * ldb + (n0 + ns0);
    bg1 = Bb + (long long)kl1 * ldb + (n0 + ns1);
    b_st0 = &Bt[ns0][kl0];                // scatter store, stride 40
    b_st1 = &Bt[ns1][kl1];
    bstep = 32 * ldb;
  } else {
    const int i1 = 256 + tid;
    const int nl0 = tid >> 2, ks0 = (tid & 3) * 8;
    const int nl1 = i1 >> 2,  ks1 = (i1 & 3) * 8;
    bg0 = Bb + (long long)(n0 + nl0) * ldb + ks0;
    bg1 = Bb + (long long)(n0 + nl1) * ldb + ks1;
    b_st0 = &Bt[nl0][ks0];                // contiguous 16B store
    b_st1 = &Bt[nl1][ks1];
    bstep = 32;
  }

  v8f acc[2][2];
  #pragma unroll
  for (int tm = 0; tm < 2; ++tm)
    #pragma unroll
    for (int tn = 0; tn < 2; ++tn)
      #pragma unroll
      for (int e = 0; e < 8; ++e) acc[tm][tn][e] = 0.0f;

  for (int k0 = 0; k0 < K; k0 += 32) {
    // ---- stage tiles: 16-byte global loads, pointer-increment addressing ----
    v8bf av  = *(const v8bf*)ag;
    v8bf bv0 = *(const v8bf*)bg0;
    v8bf bv1 = *(const v8bf*)bg1;
    ag += astep; bg0 += bstep; bg1 += bstep;
    if (k0 + 32 < K) {
      __builtin_prefetch(ag, 0, 1);
      __builtin_prefetch(bg0, 0, 1);
      __builtin_prefetch(bg1, 0, 1);
    }
    if constexpr (!ATR) {
      *(v8bf*)a_st = av;
    } else {
      #pragma unroll
      for (int e = 0; e < 8; ++e) a_st[e * 40] = av[e];
    }
    if constexpr (!BTR) {
      #pragma unroll
      for (int e = 0; e < 8; ++e) b_st0[e * 40] = bv0[e];
      #pragma unroll
      for (int e = 0; e < 8; ++e) b_st1[e * 40] = bv1[e];
    } else {
      *(v8bf*)b_st0 = bv0;
      *(v8bf*)b_st1 = bv1;
    }
    __syncthreads();

    // ---- fragments: two ds_load_b128 per fragment ----
    v16bf afr[2], bfr[2];
    #pragma unroll
    for (int tm = 0; tm < 2; ++tm) {
      int mrow = wm * 32 + tm * 16 + r16;
      v8bf lo = *(const v8bf*)&As[mrow][half * 8];
      v8bf hi = *(const v8bf*)&As[mrow][16 + half * 8];
      afr[tm] = __builtin_shufflevector(lo, hi, 0, 1, 2, 3, 4, 5, 6, 7,
                                        8, 9, 10, 11, 12, 13, 14, 15);
    }
    #pragma unroll
    for (int tn = 0; tn < 2; ++tn) {
      int ncol = wn * 32 + tn * 16 + r16;
      v8bf lo = *(const v8bf*)&Bt[ncol][half * 8];
      v8bf hi = *(const v8bf*)&Bt[ncol][16 + half * 8];
      bfr[tn] = __builtin_shufflevector(lo, hi, 0, 1, 2, 3, 4, 5, 6, 7,
                                        8, 9, 10, 11, 12, 13, 14, 15);
    }
    #pragma unroll
    for (int tm = 0; tm < 2; ++tm)
      #pragma unroll
      for (int tn = 0; tn < 2; ++tn)
        acc[tm][tn] = __builtin_amdgcn_wmma_f32_16x16x32_bf16(
            false, afr[tm], false, bfr[tn], (short)0, acc[tm][tn], false, false);
    __syncthreads();
  }

  // ---- epilogue ----
  const float alpha = alphaPtr ? alphaPtr[0] : 1.0f;
  float* Db = D + (long long)b * sD;
  bf16_t* Dbfb = Dbf ? Dbf + (long long)b * sD : (bf16_t*)0;
  const float* brow = biasRow ? biasRow + (long long)b * sBR : (const float*)0;
  const float* bcol = biasCol ? biasCol + (long long)b * sBC : (const float*)0;
  const float* rs   = rowScale ? rowScale + (long long)b * sRS : (const float*)0;

  #pragma unroll
  for (int tm = 0; tm < 2; ++tm)
    #pragma unroll
    for (int tn = 0; tn < 2; ++tn)
      #pragma unroll
      for (int v = 0; v < 8; ++v) {
        int mrow = m0 + wm * 32 + tm * 16 + half * 8 + v;
        int ncol = n0 + wn * 32 + tn * 16 + r16;
        float val = acc[tm][tn][v];
        if (brow) val += brow[mrow];
        if (bcol) val += bcol[ncol];
        if (rs)   val *= rs[mrow];
        val *= alpha;
        long long o = (long long)mrow * ldd + ncol;
        Db[o] = val;
        if (Dbfb) Dbfb[o] = (bf16_t)val;
      }
}

// ---------------------------------------------------------------------------
// Fused IDWT + (B,C,N)->(B,N,C) transpose:
//   out[b][p][c] = x4[b][c][p] + 0.5*(attT - llT)[b][pix(p)][c]
// Tiled through LDS: 32 channels x 64 pixels per block.
// ---------------------------------------------------------------------------
__global__ __launch_bounds__(256) void idwt_out_kernel(
    const float* __restrict__ x, const float* __restrict__ attT,
    const float* __restrict__ llT, float* __restrict__ out) {
  __shared__ float tile[32][65];
  const int b  = blockIdx.z;
  const int p0 = blockIdx.x * 64;
  const int c0 = blockIdx.y * 32;
  const int tid = threadIdx.x;
  const float* xb = x + ((long long)b << 23);

  #pragma unroll
  for (int it = 0; it < 8; ++it) {
    int idx = it * 256 + tid;
    int cl = idx >> 6, pl = idx & 63;
    tile[cl][pl] = xb[(long long)(c0 + cl) * NN + (p0 + pl)];
  }
  __syncthreads();
  const long long boff = ((long long)b << 21);
  #pragma unroll
  for (int it = 0; it < 8; ++it) {
    int idx = it * 256 + tid;
    int pl = idx >> 5, cl = idx & 31;
    int p = p0 + pl, c = c0 + cl;
    int w = p >> 7, h = p & 127;
    int pix = ((w >> 1) << 6) + (h >> 1);
    long long o2 = boff + ((long long)pix << 9) + c;
    float val = tile[cl][pl] + 0.5f * (attT[o2] - llT[o2]);
    out[((long long)b << 23) + ((long long)p << 9) + c] = val;
  }
}

// ---------------------------------------------------------------------------
// Host orchestration
// ---------------------------------------------------------------------------
extern "C" void kernel_launch(void* const* d_in, const int* in_sizes, int n_in,
                              void* d_out, int out_size, void* d_ws, size_t ws_size,
                              hipStream_t stream) {
  const float* x       = (const float*)d_in[0];
  const float* y       = (const float*)d_in[1];
  const float* gamma   = (const float*)d_in[2];
  const float* gamma_y = (const float*)d_in[3];
  const float* wq  = (const float*)d_in[4];  const float* bq  = (const float*)d_in[5];
  const float* wk  = (const float*)d_in[6];  const float* bk  = (const float*)d_in[7];
  const float* wv  = (const float*)d_in[8];  const float* bv  = (const float*)d_in[9];
  const float* wqy = (const float*)d_in[10]; const float* bqy = (const float*)d_in[11];
  const float* wky = (const float*)d_in[12]; const float* bky = (const float*)d_in[13];
  const float* wvy = (const float*)d_in[14]; const float* bvy = (const float*)d_in[15];

  float* outx = (float*)d_out;
  float* outy = outx + (long long)BB * NN * CC;

  // --- workspace carving ---
  size_t off = 0;
  char* wsb = (char*)d_ws;
  auto alloc = [&](size_t bytes) -> void* {
    off = (off + 255) & ~(size_t)255;
    void* p = wsb + off;
    off += bytes;
    return p;
  };
  const long long nCN = (long long)BB * CC * NP;   // 8,388,608
  const long long nMN = (long long)BB * MM * NP;   // 1,048,576

  bf16_t* w_bf[6];
  const float* w_src[6] = {wq, wk, wv, wqy, wky, wvy};
  const int    w_cnt[6] = {MM * CC, MM * CC, CC * CC, MM * CC, MM * CC, CC * CC};
  for (int i = 0; i < 6; ++i) w_bf[i] = (bf16_t*)alloc((size_t)w_cnt[i] * 2);

  struct Stream {
    bf16_t *xl_bf, *Qbf, *Kbf, *Vbf, *matBf;
    float  *llT, *Qraw, *Kraw, *Vf, *Ksum, *vsum, *tailor, *matF, *attT;
  } S[2];
  for (int s = 0; s < 2; ++s) {
    S[s].xl_bf  = (bf16_t*)alloc((size_t)nCN * 2);
    S[s].llT    = (float*)alloc((size_t)nCN * 4);
    S[s].Qraw   = (float*)alloc((size_t)nMN * 4);
    S[s].Kraw   = (float*)alloc((size_t)nMN * 4);
    S[s].Qbf    = (bf16_t*)alloc((size_t)nMN * 2);
    S[s].Kbf    = (bf16_t*)alloc((size_t)nMN * 2);
    S[s].Vf     = (float*)alloc((size_t)nCN * 4);
    S[s].Vbf    = (bf16_t*)alloc((size_t)nCN * 2);
    S[s].Ksum   = (float*)alloc((size_t)BB * MM * 4);
    S[s].vsum   = (float*)alloc((size_t)BB * CC * 4);
    S[s].tailor = (float*)alloc((size_t)BB * NP * 4);
    S[s].matF   = (float*)alloc((size_t)BB * MM * CC * 4);
    S[s].matBf  = (bf16_t*)alloc((size_t)BB * MM * CC * 2);
    S[s].attT   = (float*)alloc((size_t)nCN * 4);
  }

  // --- weight conversion ---
  for (int i = 0; i < 6; ++i)
    f32_to_bf16_kernel<<<(w_cnt[i] + 255) / 256, 256, 0, stream>>>(
        w_src[i], w_bf[i], w_cnt[i]);

  const float* inp[2]   = {x, y};
  const float* gam[2]   = {gamma, gamma_y};
  float*       outp[2]  = {outx, outy};
  const bf16_t* Wq[2]   = {w_bf[0], w_bf[3]};
  const bf16_t* Wk[2]   = {w_bf[1], w_bf[4]};
  const bf16_t* Wv[2]   = {w_bf[2], w_bf[5]};
  const float* Bq[2]    = {bq, bqy};
  const float* Bk[2]    = {bk, bky};
  const float* Bv[2]    = {bv, bvy};

  for (int s = 0; s < 2; ++s) {
    Stream& st = S[s];

    // 1) DWT LL band
    dwt_ll_kernel<<<(unsigned)(nCN / 256), 256, 0, stream>>>(
        inp[s], st.xl_bf, st.llT);

    // 2) Q/K/V projections (M x K) @ (K x N) over batch
    dim3 gQK(NP / 128, MM / 64, BB);
    gemm_bf16_wmma<false, false><<<gQK, 256, 0, stream>>>(
        Wq[s], CC, 0, st.xl_bf, NP, (long long)CC * NP,
        st.Qraw, NP, (long long)MM * NP, (bf16_t*)0,
        Bq[s], 0, (const float*)0, 0, (const float*)0, 0, (const float*)0,
        MM, NP, CC);
    gemm_bf16_wmma<false, false><<<gQK, 256, 0, stream>>>(
        Wk[s], CC, 0, st.xl_bf, NP, (long long)CC * NP,
        st.Kraw, NP, (long long)MM * NP, (bf16_t*)0,
        Bk[s], 0, (const float*)0, 0, (const float*)0, 0, (const float*)0,
        MM, NP, CC);
    dim3 gV(NP / 128, CC / 64, BB);
    gemm_bf16_wmma<false, false><<<gV, 256, 0, stream>>>(
        Wv[s], CC, 0, st.xl_bf, NP, (long long)CC * NP,
        st.Vf, NP, (long long)CC * NP, st.Vbf,
        Bv[s], 0, (const float*)0, 0, (const float*)0, 0, (const float*)0,
        CC, NP, CC);

    // 3) column L2 norms (in place) + bf16 copies
    l2n64_kernel<<<(BB * NP) / 256, 256, 0, stream>>>(st.Qraw, st.Qbf);
    l2n64_kernel<<<(BB * NP) / 256, 256, 0, stream>>>(st.Kraw, st.Kbf);

    // 4) Ksum (+EPS) and value_sum
    rowsum_kernel<<<BB * MM, 256, 0, stream>>>(
        st.Kraw, st.Ksum, NP, 1e-6f, (long long)MM * NP, MM);
    rowsum_kernel<<<BB * CC, 256, 0, stream>>>(
        st.Vf, st.vsum, NP, 0.0f, (long long)CC * NP, CC);

    // 5) tailor
    tailor_kernel<<<(BB * NP) / 256, 256, 0, stream>>>(
        st.Qraw, st.Ksum, st.tailor);

    // 6) matrix = K @ V^T   (64 x 512, K-dim 4096); B read N-major
    dim3 gM(CC / 128, MM / 64, BB);
    gemm_bf16_wmma<false, true><<<gM, 256, 0, stream>>>(
        st.Kbf, NP, (long long)MM * NP, st.Vbf, NP, (long long)CC * NP,
        st.matF, CC, (long long)MM * CC, st.matBf,
        (const float*)0, 0, (const float*)0, 0, (const float*)0, 0,
        (const float*)0, MM, CC, NP);

    // 7) attT = gamma * tailor[pix] * (Q^T @ matrix + value_sum)   (4096 x 512)
    dim3 gA(CC / 128, NP / 64, BB);
    gemm_bf16_wmma<true, false><<<gA, 256, 0, stream>>>(
        st.Qbf, NP, (long long)MM * NP, st.matBf, CC, (long long)MM * CC,
        st.attT, CC, (long long)CC * NP, (bf16_t*)0,
        (const float*)0, 0, st.vsum, CC, st.tailor, NP, gam[s],
        NP, CC, MM);

    // 8) fused IDWT identity + transpose to (B, N, C)
    dim3 gO(NN / 64, CC / 32, BB);
    idwt_out_kernel<<<gO, 256, 0, stream>>>(inp[s], st.attT, st.llT, outp[s]);
  }
}
